// S2S_MSA_69346541961320
// MI455X (gfx1250) — compile-verified, matrix operand order checked by reading, not run on hardware
//
#include <hip/hip_runtime.h>

// ---------------------------------------------------------------------------
// Problem constants (from reference)
// ---------------------------------------------------------------------------
#define BB    64
#define HH    32
#define WWID  32
#define DIMC  448
#define C2    224
#define INNER 224
#define NTOK  (BB * HH * WWID)        // 65536 tokens / pixels
#define QKVN  672                     // q(224) + k(224) + v(224)
#define SCALE 0.1889822365046136f     // 28^-0.5

typedef _Float16 h16;
typedef __attribute__((ext_vector_type(16))) _Float16 v16h;
typedef __attribute__((ext_vector_type(8)))  _Float16 v8h;
typedef __attribute__((ext_vector_type(8)))  float    v8f;

// ---------------------------------------------------------------------------
// CDNA5 async global->LDS (ASYNCcnt family, CDNA5 ISA).
// VDST operand = LDS byte address (low 32 bits of the generic shared ptr per
// the LDS aperture rule).
// ---------------------------------------------------------------------------
__device__ __forceinline__ void async_g2l_b128(unsigned int lds_addr,
                                               const void* gaddr) {
  asm volatile("global_load_async_to_lds_b128 %0, %1, off"
               :
               : "v"(lds_addr), "v"(gaddr)
               : "memory");
}

__device__ __forceinline__ void wait_async0() {
  asm volatile("s_wait_asynccnt 0" ::: "memory");
}

// ---------------------------------------------------------------------------
// WMMA helpers (CDNA5 v_wmma_f32_16x16x32_f16)
//
// A-frag (16x32 f16, row-major source, leading dim `ld` halves):
//   lane = m + 16*((k/8)&1); VGPR0-3 hold k in [kq,kq+8), VGPR4-7 k in
//   [16+kq, 16+kq+8) where kq = (lane>=16)*8.  -> two contiguous b128 loads.
// B-frag of a (32x16) matrix is loaded as the A-frag of its transpose, so all
// weight matrices are stored pre-transposed [N][K].
// ---------------------------------------------------------------------------
__device__ __forceinline__ v16h frag_ld(const h16* base, int ld) {
  const int lane = threadIdx.x & 31;
  const int r    = lane & 15;
  const int kq   = (lane >> 4) << 3;
  const h16* p = base + r * ld + kq;
  v8h lo = *(const v8h*)(p);
  v8h hi = *(const v8h*)(p + 16);
  v16h f;
#pragma unroll
  for (int i = 0; i < 8; ++i) { f[i] = lo[i]; f[i + 8] = hi[i]; }
  return f;
}

__device__ __forceinline__ v8f wmma16(v16h a, v16h b, v8f c) {
  return __builtin_amdgcn_wmma_f32_16x16x32_f16(false, a, false, b,
                                                (short)0, c, false, false);
}

__device__ __forceinline__ v8f zero8() {
  v8f z = {0.f, 0.f, 0.f, 0.f, 0.f, 0.f, 0.f, 0.f};
  return z;
}

// ---------------------------------------------------------------------------
// Weight packing (fp32 -> f16, transpose to [N][K])
// ---------------------------------------------------------------------------
__global__ void pack_wqkv(const float* __restrict__ Wq,
                          const float* __restrict__ Wkv,
                          h16* __restrict__ dst) {      // [672][224]
  int idx = blockIdx.x * 256 + threadIdx.x;
  if (idx >= QKVN * C2) return;
  int n = idx / C2, k = idx % C2;
  float v = (n < INNER) ? Wq[k * INNER + n] : Wkv[k * (2 * INNER) + (n - INNER)];
  dst[n * C2 + k] = (h16)v;
}

__global__ void pack_wout(const float* __restrict__ Wout,
                          h16* __restrict__ dst) {      // [448][224]
  int idx = blockIdx.x * 256 + threadIdx.x;
  if (idx >= DIMC * INNER) return;
  int n = idx / INNER, k = idx % INNER;
  dst[n * INNER + k] = (h16)Wout[k * DIMC + n];
}

// conv weight (Cout,Cin,3,3,3): depth=1 input => only middle depth slice.
// Repack as [kyx(9)][co(224)][ci(224)] f16.
__global__ void pack_wconv(const float* __restrict__ src,
                           h16* __restrict__ dst) {
  int idx = blockIdx.x * 256 + threadIdx.x;
  if (idx >= 9 * C2 * C2) return;
  int kyx = idx / (C2 * C2);
  int r   = idx % (C2 * C2);
  int co  = r / C2, ci = r % C2;
  dst[idx] = (h16)src[(size_t)co * (C2 * 27) + (size_t)ci * 27 + 9 + kyx];
}

// ---------------------------------------------------------------------------
// Generic GEMM: C[M=65536, 224-col group] = A[M,224] * BT^T (+bias)
// grid = (M/32, Ngroups); block = 224 threads (7 waves, 32 cols each).
// A is fp32 (stride lda, convert in VALU) or f16 (contiguous: async->LDS).
// ---------------------------------------------------------------------------
__global__ __launch_bounds__(224) void gemm224(
    const float* __restrict__ Af32, const h16* __restrict__ Af16, int lda,
    const h16* __restrict__ BT,     const float* __restrict__ bias,
    float* __restrict__ Cf32, h16* __restrict__ Cf16, int ldc) {
  __shared__ h16 At[32 * 224];

  const int m0    = blockIdx.x * 32;
  const int nbase = blockIdx.y * 224;

  if (Af16) {
    // A tile is 32 rows x 448B = 14336 contiguous bytes: stream async to LDS.
    const char* g = (const char*)(Af16 + (size_t)m0 * 224);
    const unsigned int l0 = (unsigned int)(uintptr_t)(&At[0]);
    for (int t = threadIdx.x; t < 896; t += 224)
      async_g2l_b128(l0 + t * 16, g + t * 16);
    wait_async0();
  } else {
    for (int idx = threadIdx.x; idx < 32 * 224; idx += 224) {
      int r = idx / 224, c = idx - r * 224;
      At[idx] = (h16)Af32[(size_t)(m0 + r) * lda + c];
    }
  }
  __syncthreads();

  const int wid  = threadIdx.x >> 5;
  const int lane = threadIdx.x & 31;
  const int cb   = nbase + wid * 32;

  v8f acc00 = zero8(), acc01 = zero8(), acc10 = zero8(), acc11 = zero8();

  for (int kc = 0; kc < 7; ++kc) {
    __builtin_prefetch(BT + (size_t)cb * 224 + ((kc + 1) % 7) * 32, 0, 0);
    v16h a0 = frag_ld(At + kc * 32, 224);
    v16h a1 = frag_ld(At + 16 * 224 + kc * 32, 224);
    v16h b0 = frag_ld(BT + (size_t)cb * 224 + kc * 32, 224);
    v16h b1 = frag_ld(BT + (size_t)(cb + 16) * 224 + kc * 32, 224);
    acc00 = wmma16(a0, b0, acc00);
    acc01 = wmma16(a0, b1, acc01);
    acc10 = wmma16(a1, b0, acc10);
    acc11 = wmma16(a1, b1, acc11);
  }

  v8f accs[2][2] = {{acc00, acc01}, {acc10, acc11}};
#pragma unroll
  for (int mt = 0; mt < 2; ++mt)
#pragma unroll
    for (int nt = 0; nt < 2; ++nt)
#pragma unroll
      for (int i = 0; i < 8; ++i) {
        int row = m0 + mt * 16 + ((lane >> 4) << 3) + i;
        int col = cb + nt * 16 + (lane & 15);
        float v = accs[mt][nt][i];
        if (bias) v += bias[col];
        if (Cf32) Cf32[(size_t)row * ldc + col] = v;
        else      Cf16[(size_t)row * ldc + col] = (h16)v;
      }
}

// ---------------------------------------------------------------------------
// Implicit-GEMM 3x3 conv, 224->224 ch, NHWC, pad 1, fused bias+ReLU(+res).
// grid = B*H (one image row per workgroup); block = 224 threads (7 waves).
// LDS: 3-row x 34-col halo patch of input (f16).  K = 9 * 7 chunks of 32.
// f16 input rows are contiguous (14336B) -> async global->LDS streaming.
// ---------------------------------------------------------------------------
__global__ __launch_bounds__(224) void conv3x3(
    const float* __restrict__ xf32,   // NULL or x (stride 448, ch off 224)
    const h16*  __restrict__ xf16,    // NULL or f16 [NTOK,224]
    const h16*  __restrict__ w,       // [9][224][224]
    const float* __restrict__ bias,
    const h16*  __restrict__ res,     // optional residual (added post-ReLU)
    h16*  __restrict__ of16,          // f16 out [NTOK,224] (or NULL)
    float* __restrict__ of32) {       // fp32 out (stride 672, ch off 448)
  __shared__ h16 patch[3 * 34 * 224];

  const int b = blockIdx.x >> 5;
  const int y = blockIdx.x & 31;

  if (xf16) {
    // zero halo columns (xx = -1, 32) and fully-OOB rows via DS stores
    for (int idx = threadIdx.x; idx < 3 * 34 * 224; idx += 224) {
      int ry  = idx / (34 * 224);
      int rem = idx - ry * (34 * 224);
      int xx  = rem / 224 - 1;
      int iy  = y + ry - 1;
      if (!(iy >= 0 && iy < HH && xx >= 0 && xx < WWID)) patch[idx] = (h16)0.f;
    }
    // interior rows: one contiguous 14336B async copy per valid image row
    for (int t = threadIdx.x; t < 3 * 896; t += 224) {
      int ry = t / 896, ch = t - ry * 896;   // ry uniform per iteration
      int iy = y + ry - 1;
      if (iy >= 0 && iy < HH) {
        const char* g =
            (const char*)(xf16 + ((size_t)b * HH + iy) * WWID * 224) + ch * 16;
        unsigned int l =
            (unsigned int)(uintptr_t)(patch + (ry * 34 + 1) * 224) + ch * 16;
        async_g2l_b128(l, g);
      }
    }
    wait_async0();
  } else {
    for (int idx = threadIdx.x; idx < 3 * 34 * 224; idx += 224) {
      int ry  = idx / (34 * 224);
      int rem = idx - ry * (34 * 224);
      int xx  = rem / 224 - 1;
      int ci  = rem % 224;
      int iy  = y + ry - 1;
      float v = 0.f;
      if (iy >= 0 && iy < HH && xx >= 0 && xx < WWID) {
        size_t pix = ((size_t)b * HH + iy) * WWID + xx;
        v = xf32[pix * DIMC + C2 + ci];
      }
      patch[idx] = (h16)v;
    }
  }
  __syncthreads();

  const int wid  = threadIdx.x >> 5;
  const int lane = threadIdx.x & 31;
  const int cob  = wid * 32;

  v8f acc00 = zero8(), acc01 = zero8(), acc10 = zero8(), acc11 = zero8();

  for (int kyx = 0; kyx < 9; ++kyx) {
    const int ky = kyx / 3, kx = kyx % 3;
    const h16* abase = patch + (ky * 34 + kx) * 224;
    const h16* wbase = w + (size_t)kyx * C2 * C2;
    for (int kc = 0; kc < 7; ++kc) {
      __builtin_prefetch(wbase + (size_t)cob * 224 + ((kc + 1) % 7) * 32, 0, 0);
      v16h a0 = frag_ld(abase + kc * 32, 224);
      v16h a1 = frag_ld(abase + 16 * 224 + kc * 32, 224);
      v16h b0 = frag_ld(wbase + (size_t)cob * 224 + kc * 32, 224);
      v16h b1 = frag_ld(wbase + (size_t)(cob + 16) * 224 + kc * 32, 224);
      acc00 = wmma16(a0, b0, acc00);
      acc01 = wmma16(a0, b1, acc01);
      acc10 = wmma16(a1, b0, acc10);
      acc11 = wmma16(a1, b1, acc11);
    }
  }

  v8f accs[2][2] = {{acc00, acc01}, {acc10, acc11}};
#pragma unroll
  for (int mt = 0; mt < 2; ++mt)
#pragma unroll
    for (int nt = 0; nt < 2; ++nt)
#pragma unroll
      for (int i = 0; i < 8; ++i) {
        int px = mt * 16 + ((lane >> 4) << 3) + i;      // pixel x in row
        int co = cob + nt * 16 + (lane & 15);
        float v = accs[mt][nt][i] + bias[co];
        v = fmaxf(v, 0.f);
        size_t pix = ((size_t)b * HH + y) * WWID + px;
        if (res) v += (float)res[pix * C2 + co];
        if (of16) of16[pix * C2 + co] = (h16)v;
        if (of32) of32[pix * (DIMC + C2) + DIMC + co] = v;
      }
}

// ---------------------------------------------------------------------------
// Attention half-1: 8x8 windows, seq 64, 4 heads of d=28 (padded to 32).
// grid = B*16*2 (window x head-pair); block 64 (2 waves, 1 head each).
// ---------------------------------------------------------------------------
__global__ __launch_bounds__(64) void attn1(const h16* __restrict__ qkv,
                                            const float* __restrict__ pos1,
                                            h16* __restrict__ aout) {
  __shared__ h16  Qs[2][64][32];
  __shared__ h16  Ks[2][64][32];
  __shared__ h16  Vt[2][32][64];
  __shared__ float Ss[2][64][64];

  const int hpair = (blockIdx.x & 1) * 2;
  const int bn    = blockIdx.x >> 1;
  const int b     = bn >> 4;
  const int n     = bn & 15;
  const int wy    = n >> 2, wx = n & 3;

  for (int idx = threadIdx.x; idx < 2 * 64 * 32; idx += 64) {
    int lh  = idx >> 11;
    int rem = idx & 2047;
    int i   = rem >> 5, c = rem & 31;
    int h   = hpair + lh;
    size_t tok = ((size_t)b * HH + wy * 8 + (i >> 3)) * WWID + wx * 8 + (i & 7);
    const h16* base = qkv + tok * QKVN;
    h16 qv = (h16)0.f, kv = (h16)0.f, vv = (h16)0.f;
    if (c < 28) {
      qv = base[h * 28 + c];
      kv = base[224 + h * 28 + c];
      vv = base[448 + h * 28 + c];
    }
    Qs[lh][i][c] = qv;
    Ks[lh][i][c] = kv;
    Vt[lh][c][i] = vv;
  }
  __syncthreads();

  const int wid  = threadIdx.x >> 5;
  const int lane = threadIdx.x & 31;
  const int h    = hpair + wid;

  // S = scale * Q K^T + pos
#pragma unroll
  for (int mt = 0; mt < 4; ++mt)
#pragma unroll
    for (int nt = 0; nt < 4; ++nt) {
      v16h a = frag_ld(&Qs[wid][mt * 16][0], 32);
      v16h bb = frag_ld(&Ks[wid][nt * 16][0], 32);
      v8f c = wmma16(a, bb, zero8());
#pragma unroll
      for (int i = 0; i < 8; ++i) {
        int row = mt * 16 + ((lane >> 4) << 3) + i;
        int col = nt * 16 + (lane & 15);
        Ss[wid][row][col] =
            c[i] * SCALE + pos1[((size_t)h * 64 + row) * 64 + col];
      }
    }
  __syncthreads();

  // row softmax, f16 P written in place over Ss (row stride 128 halves)
  for (int r = threadIdx.x; r < 2 * 64; r += 64) {
    int lh = r >> 6, i = r & 63;
    float* srow = &Ss[lh][i][0];
    float mx = -1e30f;
    for (int j = 0; j < 64; ++j) mx = fmaxf(mx, srow[j]);
    h16* prow = (h16*)srow;
    float sum = 0.f;
    for (int j = 0; j < 64; ++j) {
      float e = __expf(srow[j] - mx);
      sum += e;
      prow[j] = (h16)e;     // byte 2j overwrites float j/2 (already consumed)
    }
    float inv = 1.f / sum;
    for (int j = 0; j < 64; ++j) prow[j] = (h16)((float)prow[j] * inv);
  }
  __syncthreads();

  // O = P V  (P: 64x64 f16 overlay, ld = 128 halves)
  const h16* P = (const h16*)&Ss[wid][0][0];
#pragma unroll
  for (int mt = 0; mt < 4; ++mt)
#pragma unroll
    for (int nt = 0; nt < 2; ++nt) {
      v8f c = zero8();
      c = wmma16(frag_ld(P + mt * 16 * 128, 128),
                 frag_ld(&Vt[wid][nt * 16][0], 64), c);
      c = wmma16(frag_ld(P + mt * 16 * 128 + 32, 128),
                 frag_ld(&Vt[wid][nt * 16][32], 64), c);
#pragma unroll
      for (int i = 0; i < 8; ++i) {
        int row = mt * 16 + ((lane >> 4) << 3) + i;
        int d   = nt * 16 + (lane & 15);
        if (d < 28) {
          size_t tok =
              ((size_t)b * HH + wy * 8 + (row >> 3)) * WWID + wx * 8 + (row & 7);
          aout[tok * INNER + h * 28 + d] = (h16)c[i];
        }
      }
    }
}

// ---------------------------------------------------------------------------
// Attention half-2: seq = 16 windows per in-window position, 4 heads d=28.
// grid = B*64 (batch x position); block 128 (4 waves, 1 head each).
// ---------------------------------------------------------------------------
__global__ __launch_bounds__(128) void attn2(const h16* __restrict__ qkv,
                                             const float* __restrict__ pos2,
                                             h16* __restrict__ aout) {
  __shared__ h16  Qs[4][16][32];
  __shared__ h16  Ks[4][16][32];
  __shared__ h16  Vt[4][32][32];
  __shared__ float Ss[4][16][16];
  __shared__ h16  Ps[4][16][32];

  const int b  = blockIdx.x >> 6;
  const int m  = blockIdx.x & 63;
  const int my = m >> 3, mx = m & 7;

  for (int idx = threadIdx.x; idx < 4 * 16 * 32; idx += 128) {
    int h = idx >> 9;
    int rem = idx & 511;
    int i = rem >> 5, c = rem & 31;          // i = window index
    size_t tok = ((size_t)b * HH + (i >> 2) * 8 + my) * WWID + (i & 3) * 8 + mx;
    const h16* base = qkv + tok * QKVN;
    h16 qv = (h16)0.f, kv = (h16)0.f;
    if (c < 28) {
      qv = base[112 + h * 28 + c];
      kv = base[224 + 112 + h * 28 + c];
    }
    Qs[h][i][c] = qv;
    Ks[h][i][c] = kv;
  }
  for (int idx = threadIdx.x; idx < 4 * 32 * 32; idx += 128) {
    int h = idx >> 10;
    int c = (idx >> 5) & 31;
    int j = idx & 31;
    h16 vv = (h16)0.f;
    if (c < 28 && j < 16) {
      size_t tok = ((size_t)b * HH + (j >> 2) * 8 + my) * WWID + (j & 3) * 8 + mx;
      vv = qkv[tok * QKVN + 448 + 112 + h * 28 + c];
    }
    Vt[h][c][j] = vv;
  }
  __syncthreads();

  const int wid  = threadIdx.x >> 5;
  const int lane = threadIdx.x & 31;

  {
    v8f c = wmma16(frag_ld(&Qs[wid][0][0], 32), frag_ld(&Ks[wid][0][0], 32),
                   zero8());
#pragma unroll
    for (int i = 0; i < 8; ++i) {
      int row = ((lane >> 4) << 3) + i;
      int col = lane & 15;
      Ss[wid][row][col] =
          c[i] * SCALE + pos2[((size_t)wid * 16 + row) * 16 + col];
    }
  }
  __syncthreads();

  for (int r = threadIdx.x; r < 64; r += 128) {
    int h = r >> 4, i = r & 15;
    float mx2 = -1e30f;
    for (int j = 0; j < 16; ++j) mx2 = fmaxf(mx2, Ss[h][i][j]);
    float sum = 0.f;
    float e[16];
    for (int j = 0; j < 16; ++j) { e[j] = __expf(Ss[h][i][j] - mx2); sum += e[j]; }
    float inv = 1.f / sum;
    for (int j = 0; j < 16; ++j) Ps[h][i][j] = (h16)(e[j] * inv);
    for (int j = 16; j < 32; ++j) Ps[h][i][j] = (h16)0.f;
  }
  __syncthreads();

#pragma unroll
  for (int nt = 0; nt < 2; ++nt) {
    v8f c = wmma16(frag_ld(&Ps[wid][0][0], 32), frag_ld(&Vt[wid][nt * 16][0], 32),
                   zero8());
#pragma unroll
    for (int i = 0; i < 8; ++i) {
      int row = ((lane >> 4) << 3) + i;        // window index
      int d   = nt * 16 + (lane & 15);
      if (d < 28) {
        size_t tok =
            ((size_t)b * HH + (row >> 2) * 8 + my) * WWID + (row & 3) * 8 + mx;
        aout[tok * INNER + 112 + wid * 28 + d] = (h16)c[i];
      }
    }
  }
}

// ---------------------------------------------------------------------------
// Host-side launcher
// ---------------------------------------------------------------------------
extern "C" void kernel_launch(void* const* d_in, const int* in_sizes, int n_in,
                              void* d_out, int out_size, void* d_ws,
                              size_t ws_size, hipStream_t stream) {
  (void)in_sizes; (void)n_in; (void)out_size; (void)ws_size;

  const float* x    = (const float*)d_in[0];
  const float* pos1 = (const float*)d_in[1];
  const float* pos2 = (const float*)d_in[2];
  const float* Wq   = (const float*)d_in[3];
  const float* Wkv  = (const float*)d_in[4];
  const float* Wout = (const float*)d_in[5];
  const float* bout = (const float*)d_in[6];
  const float* cw[4] = {(const float*)d_in[7],  (const float*)d_in[9],
                        (const float*)d_in[11], (const float*)d_in[13]};
  const float* cb[4] = {(const float*)d_in[8],  (const float*)d_in[10],
                        (const float*)d_in[12], (const float*)d_in[14]};
  float* out = (float*)d_out;

  char* ws = (char*)d_ws;
  size_t off = 0;
  auto alloc = [&](size_t bytes) -> void* {
    void* p = ws + off;
    off += (bytes + 255) & ~(size_t)255;
    return p;
  };
  h16* WqkvT = (h16*)alloc((size_t)QKVN * C2 * 2);
  h16* WoutT = (h16*)alloc((size_t)DIMC * INNER * 2);
  h16* wconv = (h16*)alloc((size_t)4 * 9 * C2 * C2 * 2);
  h16* qkv   = (h16*)alloc((size_t)NTOK * QKVN * 2);
  h16* aout  = (h16*)alloc((size_t)NTOK * INNER * 2);
  h16* res   = (h16*)alloc((size_t)NTOK * C2 * 2);
  h16* o0    = (h16*)alloc((size_t)NTOK * C2 * 2);
  h16* o1    = (h16*)alloc((size_t)NTOK * C2 * 2);

  // weight prep
  pack_wqkv<<<(QKVN * C2 + 255) / 256, 256, 0, stream>>>(Wq, Wkv, WqkvT);
  pack_wout<<<(DIMC * INNER + 255) / 256, 256, 0, stream>>>(Wout, WoutT);
  for (int i = 0; i < 4; ++i)
    pack_wconv<<<(9 * C2 * C2 + 255) / 256, 256, 0, stream>>>(
        cw[i], wconv + (size_t)i * 9 * C2 * C2);

  // attention path
  gemm224<<<dim3(NTOK / 32, 3), 224, 0, stream>>>(x, nullptr, DIMC, WqkvT,
                                                  nullptr, nullptr, qkv, QKVN);
  attn1<<<BB * 16 * 2, 64, 0, stream>>>(qkv, pos1, aout);
  attn2<<<BB * 64, 128, 0, stream>>>(qkv, pos2, aout);
  gemm224<<<dim3(NTOK / 32, 2), 224, 0, stream>>>(nullptr, aout, INNER, WoutT,
                                                  bout, out, nullptr,
                                                  DIMC + C2);

  // conv path (x2 channels 224..447): res + 3 chained convs, all ReLU
  conv3x3<<<BB * HH, 224, 0, stream>>>(x, nullptr, wconv, cb[0], nullptr, res,
                                       nullptr);
  conv3x3<<<BB * HH, 224, 0, stream>>>(x, nullptr, wconv + (size_t)1 * 9 * C2 * C2,
                                       cb[1], nullptr, o0, nullptr);
  conv3x3<<<BB * HH, 224, 0, stream>>>(nullptr, o0, wconv + (size_t)2 * 9 * C2 * C2,
                                       cb[2], nullptr, o1, nullptr);
  conv3x3<<<BB * HH, 224, 0, stream>>>(nullptr, o1, wconv + (size_t)3 * 9 * C2 * C2,
                                       cb[3], res, nullptr, out);
}
